// EncoderLayer_867583394142
// MI455X (gfx1250) — matrix operand order repaired and checked by reading, not compiled, not run on hardware
//
#include <hip/hip_runtime.h>

// ---------------------------------------------------------------------------
// Problem constants (from the reference)
// ---------------------------------------------------------------------------
#define BB 32
#define LL 512
#define DD 256
#define HH 8
#define DKK 32
#define DVV 32
#define DFF 2048
#define CONCAT (DD * HH * DVV + DD)   // 65792
#define NROW (BB * LL)                // 16384

typedef __attribute__((ext_vector_type(16))) _Float16 v16h;
typedef __attribute__((ext_vector_type(8)))  float    v8f;

// ---------------------------------------------------------------------------
// Wave-level 16x16x32 f16 WMMA step (wave32), CDNA5 operand layouts (ISA 7.12.2):
//   A 16x32 f16 : lane<16 holds row M=lane with K in {0..7,16..23};
//                 lanes 16..31 same rows, K offset by 8.
//   B 32x16 f16 : N = lane&15; element e of v16h uses the same K as A.
//   C/D 16x16 f32: N = lane&15, M = vgpr + 8*(lane>>4).
// ---------------------------------------------------------------------------
template <typename LA, typename LB>
__device__ __forceinline__ void wmma_step_f16(v8f& acc, LA loadA, LB loadB) {
  const int lane = threadIdx.x & 31;
  const int half = lane >> 4;
  const int mr   = lane & 15;
  v16h a, b;
#pragma unroll
  for (int e = 0; e < 16; ++e) {
    const int vg = e >> 1, pos = e & 1;
    const int ka = (vg < 4) ? (half * 8 + 2 * vg + pos)
                            : (16 + half * 8 + 2 * (vg - 4) + pos);
    a[e] = loadA(mr, ka);
    b[e] = loadB(ka, mr);
  }
  acc = __builtin_amdgcn_wmma_f32_16x16x32_f16(
      false, a, false, b, (short)0, acc, false, false);
}

template <typename LA, typename LB>
__device__ __forceinline__ v8f wmma_tile_f16(int ksteps, LA loadA, LB loadB) {
  v8f acc = {};
  for (int ks = 0; ks < ksteps; ++ks) {
    const int k0 = ks << 5;
    wmma_step_f16(acc,
        [&](int m, int k) -> _Float16 { return loadA(m, k0 + k); },
        [&](int k, int n) -> _Float16 { return loadB(k0 + k, n); });
  }
  return acc;
}

template <typename ST>
__device__ __forceinline__ void store_tile_f32(const v8f& acc, ST st) {
  const int lane = threadIdx.x & 31;
  const int half = lane >> 4;
  const int nc   = lane & 15;
#pragma unroll
  for (int v = 0; v < 8; ++v) st(v + half * 8, nc, acc[v]);
}

// ---------------------------------------------------------------------------
// Tensor Data Mover: DMA a 32-row x 128-col fp32 tile of a row-major matrix
// (row stride = DFF floats) into LDS, padding +1 DWORD per 128 DWORDs so the
// LDS row stride is 129 floats (conflict-free column reads afterwards).
// Descriptor packing per CDNA5 ISA §8.3 (group0) / §8.4 (group1).
// This toolchain's builtin takes 6 args: (g0 v4u, g1 v8i, g2 v4i, g3 v4i,
// v8i extra, i32 cpol); groups 2/3/extra are zero for a 2D tensor.
// ---------------------------------------------------------------------------
#if __has_builtin(__builtin_amdgcn_tensor_load_to_lds) && \
    __has_builtin(__builtin_amdgcn_s_wait_tensorcnt)
#define HAVE_TDM 1
#endif

#ifdef HAVE_TDM
typedef __attribute__((ext_vector_type(4))) unsigned int tdm_v4u;
typedef __attribute__((ext_vector_type(8))) int          tdm_v8i;
typedef __attribute__((ext_vector_type(4))) int          tdm_v4i;

__device__ __forceinline__ void tdm_load_slab_32x128(void* lds_dst,
                                                     const float* gsrc) {
  const unsigned long long ga = (unsigned long long)gsrc;
  const unsigned int lds_off = (unsigned int)(unsigned long long)lds_dst;

  tdm_v4u g0;
  g0.x = 1u;                                      // count=1, user descriptor
  g0.y = lds_off;                                 // lds_addr (bytes)
  g0.z = (unsigned int)ga;                        // global_addr[31:0]
  g0.w = (unsigned int)(ga >> 32) | (2u << 30);   // global_addr[56:32] | type=2

  tdm_v8i g1;
  g1[0] = (int)((2u << 16)        // data_size = 4 bytes
              | (1u << 20)        // pad_enable
              | (6u << 22));      // pad_interval = 128 DWORDs (pad_amount=1 DWORD)
  g1[1] = (int)(((unsigned)DFF & 0xffffu) << 16);      // tensor_dim0 lo16 @[63:48]
  g1[2] = (int)(((unsigned)CONCAT & 0xffffu) << 16);   // tensor_dim1 lo16 @[95:80]
  g1[3] = (int)((128u << 16) | ((unsigned)CONCAT >> 16)); // tile_dim0=128 | dim1 hi16
  g1[4] = 32;                                          // tile_dim1 = 32 rows
  g1[5] = DFF;                                         // tensor_dim0_stride lo32
  g1[6] = 0;
  g1[7] = 0;

  tdm_v4i gz4 = {0, 0, 0, 0};          // groups 2/3 unused (2D tensor)
  tdm_v8i gz8 = {0, 0, 0, 0, 0, 0, 0, 0};
  __builtin_amdgcn_tensor_load_to_lds(g0, g1, gz4, gz4, gz8, 0);
}
#endif

// ---------------------------------------------------------------------------
// K1: Q/K/V projections.  (B*L,256) @ (256,256) x3, f16 outputs.
// ---------------------------------------------------------------------------
__global__ void qkv_proj_kernel(const float* __restrict__ emb,
                                const float* __restrict__ Wq,
                                const float* __restrict__ Wk,
                                const float* __restrict__ Wv,
                                _Float16* __restrict__ Qh,
                                _Float16* __restrict__ Kh,
                                _Float16* __restrict__ Vh) {
  const int gwave = blockIdx.x * 8 + (threadIdx.x >> 5);
  const int mat   = gwave / (1024 * 16);
  const int rem   = gwave % (1024 * 16);
  const int rowT  = rem >> 4;
  const int colT  = rem & 15;
  const float* W  = (mat == 0) ? Wq : (mat == 1) ? Wk : Wv;
  _Float16* O     = (mat == 0) ? Qh : (mat == 1) ? Kh : Vh;
  const float* A  = emb + (size_t)rowT * 16 * DD;

  v8f acc = wmma_tile_f16(DD / 32,
      [&](int m, int k) -> _Float16 { return (_Float16)A[m * DD + k]; },
      [&](int k, int n) -> _Float16 { return (_Float16)W[k * DD + colT * 16 + n]; });

  store_tile_f32(acc, [&](int m, int n, float v) {
    O[(size_t)(rowT * 16 + m) * DD + colT * 16 + n] = (_Float16)v;
  });
}

// ---------------------------------------------------------------------------
// K2: fused scores -> mask -> softmax -> attn (fp32 out) -> context.
// Probabilities stay in LDS for the attn@V GEMM; attn tensor written once.
// ---------------------------------------------------------------------------
__global__ void attn_fused_kernel(const _Float16* __restrict__ Qh,
                                  const _Float16* __restrict__ Kh,
                                  const _Float16* __restrict__ Vh,
                                  const unsigned char* __restrict__ mask3,
                                  float* __restrict__ attn,
                                  float* __restrict__ ctx) {
  const int qt = blockIdx.x, h = blockIdx.y, b = blockIdx.z;
  __shared__ float sc[16 * LL];
  __shared__ float red[16][17];
  const int w = threadIdx.x >> 5;
  const float scale = 0.17677669529663687f;  // 1/sqrt(DK)

  for (int kt = w; kt < LL / 16; kt += 8) {
    v8f acc = wmma_tile_f16(1,
        [&](int m, int k) -> _Float16 {
          return Qh[(size_t)(b * LL + qt * 16 + m) * DD + h * DKK + k];
        },
        [&](int k, int n) -> _Float16 {
          return Kh[(size_t)(b * LL + kt * 16 + n) * DD + h * DKK + k];
        });
    store_tile_f32(acc, [&](int m, int n, float v) {
      const int q = qt * 16 + m, key = kt * 16 + n;
      const bool msk = mask3[((size_t)b * LL + q) * LL + key] != 0;
      sc[m * LL + key] = msk ? -1.0e9f : v * scale;
    });
  }
  __syncthreads();

  const int row = threadIdx.x >> 4, sub = threadIdx.x & 15;
  float mx = -3.0e38f;
  for (int k = sub; k < LL; k += 16) mx = fmaxf(mx, sc[row * LL + k]);
  red[row][sub] = mx;
  __syncthreads();
  if (sub == 0) {
    float m2 = red[row][0];
    for (int j = 1; j < 16; ++j) m2 = fmaxf(m2, red[row][j]);
    red[row][16] = m2;
  }
  __syncthreads();
  const float rmax = red[row][16];
  float sum = 0.f;
  for (int k = sub; k < LL; k += 16) {
    const float e = __expf(sc[row * LL + k] - rmax);
    sc[row * LL + k] = e;
    sum += e;
  }
  __syncthreads();
  red[row][sub] = sum;
  __syncthreads();
  if (sub == 0) {
    float s2 = 0.f;
    for (int j = 0; j < 16; ++j) s2 += red[row][j];
    red[row][16] = s2;
  }
  __syncthreads();
  const float inv = 1.f / red[row][16];
  float* arow = attn + ((size_t)(b * HH + h) * LL + qt * 16 + row) * LL;
  for (int k = sub; k < LL; k += 16) {
    const float p = sc[row * LL + k] * inv;
    arow[k] = p;
    sc[row * LL + k] = p;
  }
  __syncthreads();

  if (w < 2) {
    v8f acc = wmma_tile_f16(LL / 32,
        [&](int m, int k) -> _Float16 { return (_Float16)sc[m * LL + k]; },
        [&](int k, int n) -> _Float16 {
          return Vh[(size_t)(b * LL + k) * DD + h * DVV + w * 16 + n];
        });
    store_tile_f32(acc, [&](int m, int n, float v) {
      ctx[(size_t)(b * LL + qt * 16 + m) * DD + h * DVV + w * 16 + n] = v;
    });
  }
}

// ---------------------------------------------------------------------------
// K3: pool-weight softmax over L (masked context) + avg_emb tail of x.
// ---------------------------------------------------------------------------
__global__ void pool_softmax_kernel(const float* __restrict__ ctx,
                                    const int* __restrict__ imask,
                                    const float* __restrict__ emb,
                                    _Float16* __restrict__ wgt,
                                    _Float16* __restrict__ xh) {
  const int b = blockIdx.x, c = threadIdx.x;
  const float* cb = ctx + (size_t)b * LL * DD;
  const float* eb = emb + (size_t)b * LL * DD;
  const int* mb = imask + b * LL;

  float mx = -3.0e38f, asum = 0.f;
  int cnt = 0;
  for (int l = 0; l < LL; ++l) {
    const float v = mb[l] ? cb[l * DD + c] : -1.0e9f;
    mx = fmaxf(mx, v);
    if (mb[l]) { asum += eb[l * DD + c]; ++cnt; }
  }
  float sum = 0.f;
  for (int l = 0; l < LL; ++l) {
    const float v = mb[l] ? cb[l * DD + c] : -1.0e9f;
    sum += __expf(v - mx);
  }
  const float inv = 1.f / sum;
  for (int l = 0; l < LL; ++l) {
    const float v = mb[l] ? cb[l * DD + c] : -1.0e9f;
    wgt[((size_t)b * LL + l) * DD + c] = (_Float16)(__expf(v - mx) * inv);
  }
  const float len = (float)(cnt < 1 ? 1 : cnt);
  xh[(size_t)b * CONCAT + DD * HH * DVV + c] = (_Float16)(asum / len);
}

// ---------------------------------------------------------------------------
// K4: pooled[b,d,c] = sum_l emb[b,l,d]*w[b,l,c]  -> head of x (f16).
// ---------------------------------------------------------------------------
__global__ void pooled_gemm_kernel(const float* __restrict__ emb,
                                   const _Float16* __restrict__ wgt,
                                   _Float16* __restrict__ xh) {
  const int cT = blockIdx.x, dT = blockIdx.y, b = blockIdx.z;
  const float* eb = emb + (size_t)b * LL * DD;
  const _Float16* wb = wgt + (size_t)b * LL * DD;

  v8f acc = wmma_tile_f16(LL / 32,
      [&](int m, int k) -> _Float16 { return (_Float16)eb[k * DD + dT * 16 + m]; },
      [&](int k, int n) -> _Float16 { return wb[k * DD + cT * 16 + n]; });

  store_tile_f32(acc, [&](int m, int n, float v) {
    xh[(size_t)b * CONCAT + (dT * 16 + m) * DD + cT * 16 + n] = (_Float16)v;
  });
}

// ---------------------------------------------------------------------------
// K5: fc1 partials, TDM double-buffered.
// Grid: (16 col-groups x 128 cols, 2 row-tiles, 8 K-chunks); block = 8 waves.
// Wave 0 DMAs the next 32x128 fp32 weight slab to LDS (tensor_load_to_lds,
// TENSORcnt) while all 8 waves run WMMA on the current slab. LDS row stride
// 129 floats (TDM pad) -> conflict-free column reads.
// ---------------------------------------------------------------------------
#define FC1_NKS (CONCAT / 32 / 8)   // 257 k-steps per K-chunk

__global__ void fc1_partial_kernel(const _Float16* __restrict__ xh,
                                   const float* __restrict__ w1,
                                   float* __restrict__ h1p) {
  const int cg = blockIdx.x;            // 128-column group
  const int rT = blockIdx.y;            // 16-row tile
  const int kc = blockIdx.z;            // K chunk
  const int w  = threadIdx.x >> 5;
  const int colbase = cg * 128;
  const int kbase   = kc * FC1_NKS * 32;

  __shared__ float slab[2][32 * 129];

  const _Float16* xb = xh + (size_t)rT * 16 * CONCAT + kbase;

  auto stage = [&](int buf, int ks) {
    const float* gsrc = w1 + (size_t)(kbase + ks * 32) * DFF + colbase;
#ifdef HAVE_TDM
    if (threadIdx.x < 32) tdm_load_slab_32x128(&slab[buf][0], gsrc);
#else
    for (int idx = threadIdx.x; idx < 32 * 128; idx += 256) {
      const int r = idx >> 7, c = idx & 127;
      slab[buf][r * 129 + c] = gsrc[(size_t)r * DFF + c];
    }
#endif
  };

  stage(0, 0);
  v8f acc = {};
  for (int ks = 0; ks < FC1_NKS; ++ks) {
    const int cur = ks & 1;
    if (ks + 1 < FC1_NKS) {
      stage(cur ^ 1, ks + 1);
#ifdef HAVE_TDM
      if (threadIdx.x < 32) __builtin_amdgcn_s_wait_tensorcnt(1);
#endif
    } else {
#ifdef HAVE_TDM
      if (threadIdx.x < 32) __builtin_amdgcn_s_wait_tensorcnt(0);
#endif
    }
    __syncthreads();   // slab[cur] visible to all waves

    const float* sb = &slab[cur][0];
    const _Float16* xk = xb + ks * 32;
    wmma_step_f16(acc,
        [&](int m, int k) -> _Float16 { return xk[(size_t)m * CONCAT + k]; },
        [&](int k, int n) -> _Float16 {
          return (_Float16)sb[k * 129 + w * 16 + n];
        });
    __syncthreads();   // done with slab[cur] before it is re-staged
  }

  store_tile_f32(acc, [&](int m, int n, float v) {
    h1p[(size_t)kc * (32 * DFF) + (rT * 16 + m) * DFF + colbase + w * 16 + n] = v;
  });
}

__global__ void fc1_reduce_kernel(const float* __restrict__ h1p,
                                  const float* __restrict__ b1,
                                  _Float16* __restrict__ h1) {
  const int i = blockIdx.x * blockDim.x + threadIdx.x;
  float s = b1[i & (DFF - 1)];
#pragma unroll
  for (int c = 0; c < 8; ++c) s += h1p[(size_t)c * (32 * DFF) + i];
  h1[i] = (_Float16)fmaxf(s, 0.f);
}

// ---------------------------------------------------------------------------
// K6: fc2  (32 x 2048) @ (2048 x 2048) + bias, relu.
// ---------------------------------------------------------------------------
__global__ void fc2_kernel(const _Float16* __restrict__ h1,
                           const float* __restrict__ w2,
                           const float* __restrict__ b2,
                           _Float16* __restrict__ h2) {
  const int cT = blockIdx.x, rT = blockIdx.y;
  v8f acc = wmma_tile_f16(DFF / 32,
      [&](int m, int k) -> _Float16 { return h1[(size_t)(rT * 16 + m) * DFF + k]; },
      [&](int k, int n) -> _Float16 { return (_Float16)w2[(size_t)k * DFF + cT * 16 + n]; });
  store_tile_f32(acc, [&](int m, int n, float v) {
    h2[(size_t)(rT * 16 + m) * DFF + cT * 16 + n] =
        (_Float16)fmaxf(v + b2[cT * 16 + n], 0.f);
  });
}

// ---------------------------------------------------------------------------
// K7: fc3  (32 x 2048) @ (2048 x 1) + bias -> out[0..31].
// ---------------------------------------------------------------------------
__global__ void fc3_kernel(const _Float16* __restrict__ h2,
                           const float* __restrict__ w3,
                           const float* __restrict__ b3,
                           float* __restrict__ out) {
  const int r = threadIdx.x;
  float s = b3[0];
  for (int j = 0; j < DFF; ++j) s += (float)h2[(size_t)r * DFF + j] * w3[j];
  out[r] = s;
}

// ---------------------------------------------------------------------------
// Host launcher.  Workspace layout (bytes), with dead-buffer reuse:
//   Qh   [0,         8388608)   f16   (later reused for wgt)
//   Kh   [8388608,  16777216)   f16
//   Vh   [16777216, 25165824)   f16
//   ctx  [25165824, 41943040)   f32
//   xh   [41943040, 46153728)   f16
//   h1p  [46153728, 48250880)   f32
//   h1   [48250880, 48381952)   f16
//   h2   [48381952, 48513024)   f16
// d_out = [ out(32 f32) | attn(B*H*L*L f32) ]
// ---------------------------------------------------------------------------
extern "C" void kernel_launch(void* const* d_in, const int* in_sizes, int n_in,
                              void* d_out, int out_size, void* d_ws, size_t ws_size,
                              hipStream_t stream) {
  (void)in_sizes; (void)n_in; (void)out_size; (void)ws_size;

  const float* emb          = (const float*)d_in[0];
  const unsigned char* msk3 = (const unsigned char*)d_in[1];  // jax bool -> 1 byte
  const int* imask          = (const int*)d_in[2];
  const float* Wq           = (const float*)d_in[3];
  const float* Wk           = (const float*)d_in[4];
  const float* Wv           = (const float*)d_in[5];
  const float* w1           = (const float*)d_in[6];
  const float* b1           = (const float*)d_in[7];
  const float* w2           = (const float*)d_in[8];
  const float* b2           = (const float*)d_in[9];
  const float* w3           = (const float*)d_in[10];
  const float* b3           = (const float*)d_in[11];

  char* ws = (char*)d_ws;
  _Float16* Qh  = (_Float16*)(ws + 0);
  _Float16* Kh  = (_Float16*)(ws + 8388608);
  _Float16* Vh  = (_Float16*)(ws + 16777216);
  float*    ctx = (float*)   (ws + 25165824);
  _Float16* wgt = (_Float16*)(ws + 0);          // reuse Qh region (dead)
  _Float16* xh  = (_Float16*)(ws + 41943040);
  float*    h1p = (float*)   (ws + 46153728);
  _Float16* h1  = (_Float16*)(ws + 48250880);
  _Float16* h2  = (_Float16*)(ws + 48381952);

  float* out  = (float*)d_out;
  float* attn = out + BB;

  qkv_proj_kernel<<<dim3(49152 / 8), 256, 0, stream>>>(emb, Wq, Wk, Wv, Qh, Kh, Vh);
  attn_fused_kernel<<<dim3(LL / 16, HH, BB), 256, 0, stream>>>(Qh, Kh, Vh, msk3, attn, ctx);
  pool_softmax_kernel<<<dim3(BB), 256, 0, stream>>>(ctx, imask, emb, wgt, xh);
  pooled_gemm_kernel<<<dim3(16, 16, BB), 32, 0, stream>>>(emb, wgt, xh);
  fc1_partial_kernel<<<dim3(16, 2, 8), 256, 0, stream>>>(xh, w1, h1p);
  fc1_reduce_kernel<<<dim3(32 * DFF / 256), 256, 0, stream>>>(h1p, b1, h1);
  fc2_kernel<<<dim3(DFF / 16, 2), 32, 0, stream>>>(h1, w2, b2, h2);
  fc3_kernel<<<1, 32, 0, stream>>>(h2, w3, b3, out);
}
